// MaskedAttention_10411000726290
// MI455X (gfx1250) — matrix operand order repaired and checked by reading, not compiled, not run on hardware
//
#include <hip/hip_runtime.h>

#define B_   16
#define M_   1024
#define N_   4096
#define CIN_ 67
#define HID_ 128
#define BN_EPS 1e-5f

typedef __attribute__((ext_vector_type(16))) _Float16 v16h;
typedef __attribute__((ext_vector_type(8)))  float    v8f;
typedef unsigned int v4u  __attribute__((ext_vector_type(4)));
typedef int          v8i_ __attribute__((ext_vector_type(8)));
typedef int          v4i_ __attribute__((ext_vector_type(4)));

#define SCORES_BYTES (16 * N_ * 4)          // 256 KB LDS score strip
#define QSM_OFFSET   SCORES_BYTES           // q tile right after scores
#define RINV_OFFSET  (SCORES_BYTES + 16 * HID_ * 2)

// ---------------------------------------------------------------------------
// Pass 0: zero the per-(b,n) column-sum accumulator (ws is poisoned 0xAA).
// ---------------------------------------------------------------------------
__global__ void ma_zero_kernel(float* __restrict__ p, int n) {
    int i = blockIdx.x * 256 + threadIdx.x;
    if (i < n) p[i] = 0.0f;
}

// ---------------------------------------------------------------------------
// Pass 1: fused 1x1-conv + BatchNorm (inference form), output f16 q/k.
// ---------------------------------------------------------------------------
__global__ __launch_bounds__(HID_)
void ma_proj_bn_kernel(const float* __restrict__ cent, const float* __restrict__ feat,
                       const float* __restrict__ Wq,   const float* __restrict__ Wk,
                       const float* __restrict__ gq, const float* __restrict__ bq,
                       const float* __restrict__ mq, const float* __restrict__ vq,
                       const float* __restrict__ gk, const float* __restrict__ bk,
                       const float* __restrict__ mk, const float* __restrict__ vk,
                       _Float16* __restrict__ qh, _Float16* __restrict__ kh)
{
    __shared__ float xs[CIN_ + 1];
    const int r = blockIdx.x;
    const float *W, *g, *bb, *mm, *vv, *src;
    _Float16* dst;
    if (r < B_ * M_) {
        src = cent + (size_t)r * CIN_;
        W = Wq; g = gq; bb = bq; mm = mq; vv = vq;
        dst = qh + (size_t)r * HID_;
    } else {
        const int r2 = r - B_ * M_;
        src = feat + (size_t)r2 * CIN_;
        W = Wk; g = gk; bb = bk; mm = mk; vv = vk;
        dst = kh + (size_t)r2 * HID_;
    }
    const int h = threadIdx.x;
    if (h < CIN_) xs[h] = src[h];
    __syncthreads();

    float acc = 0.0f;
    const float* wrow = W + h * CIN_;
    #pragma unroll 1
    for (int c = 0; c < CIN_; ++c) acc += xs[c] * wrow[c];

    const float s   = g[h] * rsqrtf(vv[h] + BN_EPS);
    const float val = acc * s + (bb[h] - mm[h] * s);
    dst[h] = (_Float16)val;
}

// ---------------------------------------------------------------------------
// Pass 2: per (b, 16-row m-tile):
//   - TDM stages the 16x128 f16 q tile into LDS (tensor_load_to_lds)
//   - WMMA f16 16x16x32 computes the 16x4096 score strip into 256 KB LDS,
//     two independent accumulator tiles per wave to hide WMMA->WMMA hazards
//   - mask tile prefetched during the GEMM (global_prefetch_b8)
//   - masked row softmax entirely in LDS
//   - t = sqrt(mask+eps)*sqrt(p+eps)-eps -> d_out, column partials -> gcol
// ---------------------------------------------------------------------------
__global__ __launch_bounds__(256)
void ma_attn_kernel(const _Float16* __restrict__ qh, const _Float16* __restrict__ kh,
                    const float* __restrict__ mask,
                    float* __restrict__ out, float* __restrict__ gcol)
{
    extern __shared__ char smem[];
    float*    scores = (float*)smem;
    _Float16* qsm    = (_Float16*)(smem + QSM_OFFSET);
    float*    rinv   = (float*)(smem + RINV_OFFSET);

    const int tid = threadIdx.x;
    const int b   = blockIdx.x >> 6;          // 64 m-tiles per batch
    const int m0  = (blockIdx.x & 63) << 4;

    // ---- stage q tile (16 x 128 halves) into LDS ----
#if __has_builtin(__builtin_amdgcn_tensor_load_to_lds) && \
    __has_builtin(__builtin_amdgcn_s_wait_tensorcnt)
    if (tid < 32) {                           // one wave issues the TDM DMA
        const unsigned long long ga =
            (unsigned long long)(const void*)(qh + ((size_t)(b * M_ + m0)) * HID_);
        // D# group 0: count=1, lds_addr, 57-bit global addr, type=2 (ISA 8.3)
        v4u g0;
        g0[0] = 1u;                                        // count=1, user mode
        g0[1] = (unsigned)QSM_OFFSET;                      // lds_addr (dyn LDS @0)
        g0[2] = (unsigned)(ga & 0xffffffffu);              // global_addr[31:0]
        g0[3] = (unsigned)((ga >> 32) & 0x01ffffffu) | (2u << 30); // [56:32]|type=2
        // D# group 1: data_size=2B; tensor 128x16; tile 128x16; stride 128 (ISA 8.4)
        v8i_ g1;
        g1[0] = (1 << 16);          // workgroup_mask=0, data_size=1 (2 bytes)
        g1[1] = (HID_ << 16);       // tensor_dim0[15:0]=128
        g1[2] = (16 << 16);         // tensor_dim0 hi=0 | tensor_dim1[15:0]=16
        g1[3] = (HID_ << 16);       // tensor_dim1 hi=0 | tile_dim0=128
        g1[4] = 16;                 // tile_dim1=16, tile_dim2=0
        g1[5] = HID_;               // tensor_dim0_stride[31:0]=128
        g1[6] = 0;                  // stride hi / tensor_dim1_stride lo = 0
        g1[7] = 0;
        v4i_ gz4 = {0, 0, 0, 0};                 // groups 2/3 unused (2D tensor)
        v8i_ gz8 = {0, 0, 0, 0, 0, 0, 0, 0};     // extra group (clang-23 6-arg form)
        __builtin_amdgcn_tensor_load_to_lds(g0, g1, gz4, gz4, gz8, 0);
        __builtin_amdgcn_s_wait_tensorcnt(0);
    }
#else
    {
        const uint32_t* qsrc = (const uint32_t*)(qh + ((size_t)(b * M_ + m0)) * HID_);
        uint32_t* qdst = (uint32_t*)qsm;
        for (int i = tid; i < 16 * HID_ / 2; i += 256) qdst[i] = qsrc[i];
    }
#endif

    // ---- prefetch the 256 KB mask tile needed by the softmax phase ----
    {
        const char* mpref = (const char*)(mask + ((size_t)(b * M_ + m0)) * N_);
        #pragma unroll 1
        for (int i = tid; i < (16 * N_ * 4) / 64; i += 256)
            __builtin_prefetch(mpref + (size_t)i * 64, 0, 1);
    }
    __syncthreads();

    // ---- WMMA: scores[16][4096] = qtile @ k^T ----
    const int lane = tid & 31;
    const int wid  = tid >> 5;                // 8 waves
    const int mrow = lane & 15;
    const int gsel = lane >> 4;

    // A fragments (16x32 f16 per K-chunk), built once per wave (ISA 7.12.2)
    v16h afrag[4];
    #pragma unroll
    for (int kk = 0; kk < 4; ++kk) {
        const int kb = kk * 32;
        #pragma unroll
        for (int e = 0; e < 8; ++e) {
            afrag[kk][e]     = qsm[mrow * HID_ + kb + 8 * gsel + e];
            afrag[kk][e + 8] = qsm[mrow * HID_ + kb + 16 + 8 * gsel + e];
        }
    }

    // two independent accumulator tiles per wave -> interleaved WMMA chains
    #pragma unroll 1
    for (int t = 0; t < (N_ / 16) / 16; ++t) {
        const int nt0 = wid + 16 * t;
        const int nt1 = nt0 + 8;
        const _Float16* krow0 = kh + ((size_t)(b * N_ + nt0 * 16 + mrow)) * HID_;
        const _Float16* krow1 = kh + ((size_t)(b * N_ + nt1 * 16 + mrow)) * HID_;
        v8f acc0 = {};
        v8f acc1 = {};
        #pragma unroll
        for (int kk = 0; kk < 4; ++kk) {
            const int kb = kk * 32;
            typedef __attribute__((ext_vector_type(8))) _Float16 v8h_;
            v8h_ b0l = *(const v8h_*)(krow0 + kb + 8 * gsel);
            v8h_ b0h = *(const v8h_*)(krow0 + kb + 16 + 8 * gsel);
            v8h_ b1l = *(const v8h_*)(krow1 + kb + 8 * gsel);
            v8h_ b1h = *(const v8h_*)(krow1 + kb + 16 + 8 * gsel);
            v16h bf0 = __builtin_shufflevector(b0l, b0h, 0,1,2,3,4,5,6,7,8,9,10,11,12,13,14,15);
            v16h bf1 = __builtin_shufflevector(b1l, b1h, 0,1,2,3,4,5,6,7,8,9,10,11,12,13,14,15);
            acc0 = __builtin_amdgcn_wmma_f32_16x16x32_f16(
                       false, afrag[kk], false, bf0, (short)0, acc0, false, false);
            acc1 = __builtin_amdgcn_wmma_f32_16x16x32_f16(
                       false, afrag[kk], false, bf1, (short)0, acc1, false, false);
        }
        #pragma unroll
        for (int rr = 0; rr < 8; ++rr) {
            scores[(rr + 8 * gsel) * N_ + nt0 * 16 + mrow] = acc0[rr];
            scores[(rr + 8 * gsel) * N_ + nt1 * 16 + mrow] = acc1[rr];
        }
    }
    __syncthreads();

    // ---- masked softmax per row (16 threads per row, shfl width-16) ----
    const int row = tid >> 4;
    const int lc  = tid & 15;
    const float* mrowp = mask + ((size_t)(b * M_ + m0 + row)) * N_;

    float pmax = -3.4e38f;
    for (int j = lc; j < N_; j += 16) {
        float v = scores[row * N_ + j];
        v = (mrowp[j] < 1e-9f) ? -1e9f : v;
        scores[row * N_ + j] = v;
        pmax = fmaxf(pmax, v);
    }
    for (int o = 8; o > 0; o >>= 1) pmax = fmaxf(pmax, __shfl_xor(pmax, o, 16));

    float psum = 0.0f;
    for (int j = lc; j < N_; j += 16) {
        float e = __expf(scores[row * N_ + j] - pmax);
        scores[row * N_ + j] = e;
        psum += e;
    }
    for (int o = 8; o > 0; o >>= 1) psum += __shfl_xor(psum, o, 16);
    if (lc == 0) rinv[row] = 1.0f / psum;
    __syncthreads();

    // ---- t = sqrt(mask+eps)*sqrt(softmax+eps)-eps; column partial sums ----
    float*       orow  = out  + ((size_t)(b * M_ + m0)) * N_;
    const float* mbase = mask + ((size_t)(b * M_ + m0)) * N_;
    for (int i = 0; i < 16; ++i) {
        const int col = tid + 256 * i;
        float csum = 0.0f;
        #pragma unroll 1
        for (int r = 0; r < 16; ++r) {
            const float p  = scores[r * N_ + col] * rinv[r];
            const float mk = mbase[(size_t)r * N_ + col];
            const float tt = sqrtf(mk + 1e-9f) * sqrtf(p + 1e-9f) - 1e-9f;
            orow[(size_t)r * N_ + col] = tt;
            csum += fabsf(tt);
        }
        atomicAdd(gcol + b * N_ + col, csum);
    }
}

// ---------------------------------------------------------------------------
// Pass 3: rowsum[b,m] = sum_n |t| / max(colsum[b,n], 1e-12)
// ---------------------------------------------------------------------------
__global__ __launch_bounds__(256)
void ma_rowsum_kernel(const float* __restrict__ out, const float* __restrict__ gcol,
                      float* __restrict__ grow)
{
    __shared__ float red[256];
    const int bm = blockIdx.x;
    const int b  = bm / M_;
    const float* trow = out + (size_t)bm * N_;
    const float* crow = gcol + b * N_;
    float p = 0.0f;
    for (int n = threadIdx.x; n < N_; n += 256)
        p += fabsf(trow[n]) / fmaxf(crow[n], 1e-12f);
    red[threadIdx.x] = p;
    __syncthreads();
    for (int s = 128; s > 0; s >>= 1) {
        if (threadIdx.x < s) red[threadIdx.x] += red[threadIdx.x + s];
        __syncthreads();
    }
    if (threadIdx.x == 0) grow[bm] = red[0];
}

// ---------------------------------------------------------------------------
// Pass 4: out = t / (max(colsum,1e-12) * max(rowsum,1e-12)), in place
// ---------------------------------------------------------------------------
__global__ __launch_bounds__(256)
void ma_scale_kernel(float* __restrict__ out, const float* __restrict__ gcol,
                     const float* __restrict__ grow)
{
    const size_t total = (size_t)B_ * M_ * N_;
    for (size_t idx = (size_t)blockIdx.x * 256 + threadIdx.x; idx < total;
         idx += (size_t)gridDim.x * 256) {
        const size_t col = idx & (N_ - 1);   // % 4096
        const size_t bm  = idx >> 12;        // / 4096
        const size_t b   = bm >> 10;         // / 1024
        const float d1 = fmaxf(gcol[b * N_ + col], 1e-12f);
        const float d2 = fmaxf(grow[bm], 1e-12f);
        out[idx] = out[idx] / (d1 * d2);
    }
}

// ---------------------------------------------------------------------------
extern "C" void kernel_launch(void* const* d_in, const int* in_sizes, int n_in,
                              void* d_out, int out_size, void* d_ws, size_t ws_size,
                              hipStream_t stream)
{
    const float* cent = (const float*)d_in[0];
    const float* feat = (const float*)d_in[1];
    const float* mask = (const float*)d_in[2];
    const float* Wq   = (const float*)d_in[3];
    const float* Wk   = (const float*)d_in[4];
    const float* gq = (const float*)d_in[5],  *bq = (const float*)d_in[6];
    const float* mq = (const float*)d_in[7],  *vq = (const float*)d_in[8];
    const float* gk = (const float*)d_in[9],  *bk = (const float*)d_in[10];
    const float* mkp = (const float*)d_in[11], *vk = (const float*)d_in[12];
    float* out = (float*)d_out;

    char* ws = (char*)d_ws;
    _Float16* qh = (_Float16*)ws;                                   //  4 MB
    _Float16* kh = (_Float16*)(ws + (size_t)B_ * M_ * HID_ * 2);    // 16 MB
    float* gcol = (float*)(ws + (size_t)B_ * M_ * HID_ * 2
                              + (size_t)B_ * N_ * HID_ * 2);        // 256 KB
    float* grow = gcol + B_ * N_;                                   //  64 KB

    ma_zero_kernel<<<(B_ * N_ + 255) / 256, 256, 0, stream>>>(gcol, B_ * N_);

    ma_proj_bn_kernel<<<B_ * (M_ + N_), HID_, 0, stream>>>(
        cent, feat, Wq, Wk, gq, bq, mq, vq, gk, bk, mkp, vk, qh, kh);

    const size_t shbytes = RINV_OFFSET + 64;  // ~260 KB dynamic LDS
    ma_attn_kernel<<<B_ * (M_ / 16), 256, shbytes, stream>>>(qh, kh, mask, out, gcol);

    ma_rowsum_kernel<<<B_ * M_, 256, 0, stream>>>(out, gcol, grow);

    ma_scale_kernel<<<8192, 256, 0, stream>>>(out, gcol, grow);
}
